// AdditiveAttention_7129645711646
// MI455X (gfx1250) — matrix operand order repaired and checked by reading, not compiled, not run on hardware
//
#include <hip/hip_runtime.h>
#include <math.h>

// ---------------- CDNA5 WMMA types ----------------
typedef __attribute__((ext_vector_type(16))) __bf16 v16bf;
typedef __attribute__((ext_vector_type(8)))  float  v8f;

#define H    512
#define SEQ  2048
#define BATCH 64
#define NEG_INF -1e10f

// LDS A-tile row stride (bf16 elems), padded to kill bank conflicts:
// 1040 B = 260 dwords, 260 % 64 = 4 -> rows land on distinct 4-bank groups,
// matching the 4-bank footprint of ds_load_b128.
#define A_ROW_BF16 520
#define A_ROW_BYTES (A_ROW_BF16 * 2)   // 1040, 16B-aligned
#define A_ROWS 64                       // rows staged per workgroup
#define DYN_LDS_BYTES (A_ROWS * A_ROW_BYTES)   // 66560

union AFrag { v16bf v; int4 q[2]; };

// round-to-nearest-even f32 -> bf16 (bit manipulation; no __bf16 arithmetic)
static __device__ inline unsigned short f2bf(float x) {
    unsigned u = __builtin_bit_cast(unsigned, x);
    unsigned r = (u + 0x7fffu + ((u >> 16) & 1u)) >> 16;
    return (unsigned short)r;
}
static __device__ inline unsigned pack2bf(float a, float b) {
    return (unsigned)f2bf(a) | ((unsigned)f2bf(b) << 16);
}

// branch-free tanh on the hw exp unit: tanh(x) = sign(x) * (1 - 2/(exp(2|x|)+1)).
// exp overflow -> inf -> 2/inf = 0 -> result +-1 (correct saturation); no exec divergence.
static __device__ inline float fast_tanh(float x) {
    float ax = fabsf(x);
    float e  = __expf(2.0f * ax);          // v_exp_f32
    float t  = 1.0f - 2.0f / (e + 1.0f);   // v_rcp_f32 + fma
    return copysignf(t, x);                // v_bfi
}

// ---------------- K1: W1 [h][d] f32 -> W1T [d][h] bf16 ----------------
__global__ void __launch_bounds__(256)
w1t_kernel(const float* __restrict__ W1, unsigned short* __restrict__ w1t) {
    int i = blockIdx.x * 256 + threadIdx.x;      // over H*H
    int h = i >> 9;
    int d = i & (H - 1);
    w1t[d * H + h] = f2bf(W1[i]);
}

// ---------------- K2: qproj[b][d] = sum_h output[b][h] * W2[h][d] ----------------
__global__ void __launch_bounds__(256)
qproj_kernel(const float* __restrict__ outp, const float* __restrict__ W2,
             float* __restrict__ qproj) {
    __shared__ float s_q[H];
    int b = blockIdx.x, tid = threadIdx.x;
    for (int i = tid; i < H; i += 256) s_q[i] = outp[b * H + i];
    __syncthreads();
    for (int d = tid; d < H; d += 256) {
        float acc = 0.f;
        for (int h = 0; h < H; ++h) acc += s_q[h] * W2[h * H + d];
        qproj[b * H + d] = acc;
    }
}

// ---------------- K3: energy[b][s] = V . tanh(enc[b,s,:]@W1 + qproj[b,:]) ----------------
// grid = B * S/64 blocks, 64 threads (2 waves, wave32).
// Each wave owns 32 rows (two 16-row A sub-tiles) -> every B fragment feeds 2 WMMAs,
// halving W1T cache traffic vs 16 rows/wave.
__global__ void __launch_bounds__(64)
energy_kernel(const float* __restrict__ enc,            // [B,S,H] f32
              const unsigned short* __restrict__ w1t,   // [H(d)][H(h)] bf16
              const float* __restrict__ qproj,          // [B,H]
              const float* __restrict__ Vvec,           // [H]
              float* __restrict__ energy)               // [B,S]
{
    extern __shared__ unsigned char smemA[];            // A_ROWS x A_ROW_BYTES (dynamic)
    __shared__ float s_qp[H];
    __shared__ float s_v[H];

    const int tid  = threadIdx.x;
    const int lane = tid & 31;
    const int wave = tid >> 5;                 // 0..1
    const int hf   = lane >> 4;                // lane half (K sub-block select)
    const int rsub = lane & 15;                // row within 16x16 tile / column idx
    const int tilesPerB = SEQ / A_ROWS;        // 32
    const int b     = blockIdx.x / tilesPerB;
    const int srow0 = (blockIdx.x % tilesPerB) * A_ROWS;

    for (int i = tid; i < H; i += 64) {
        s_qp[i] = qproj[b * H + i];
        s_v[i]  = Vvec[i];
    }
    // stage A tile: 64 rows x 512 f32 -> bf16 in LDS (padded rows)
    const float* encB = enc + ((size_t)b * SEQ + srow0) * H;
    for (int i = tid; i < A_ROWS * (H / 2); i += 64) {
        int row = i >> 8;                      // H/2 = 256 pairs per row
        int p   = i & 255;
        float2 f = ((const float2*)(encB + (size_t)row * H))[p];
        *(unsigned*)(smemA + row * A_ROW_BYTES + p * 4) = pack2bf(f.x, f.y);
    }
    __syncthreads();

    const unsigned char* aBase0 = smemA + (wave * 32 + rsub) * A_ROW_BYTES;
    const unsigned char* aBase1 = aBase0 + 16 * A_ROW_BYTES;
    const unsigned char* w1tb   = (const unsigned char*)w1t;

    float e0[8], e1[8];
#pragma unroll
    for (int r = 0; r < 8; ++r) { e0[r] = 0.f; e1[r] = 0.f; }

    for (int ng = 0; ng < 8; ++ng) {           // 8 N-blocks of 4 tiles = 512 cols
        v8f acc[2][4];
#pragma unroll
        for (int j = 0; j < 4; ++j) { acc[0][j] = (v8f){}; acc[1][j] = (v8f){}; }

#pragma unroll 2
        for (int kc = 0; kc < 16; ++kc) {      // K = 512 in chunks of 32
            const int kb = kc * 32;
            AFrag a0, a1;                      // 16-bit A layout: two b128 LDS loads each
            a0.q[0] = *(const int4*)(aBase0 + (kb + hf * 8) * 2);
            a0.q[1] = *(const int4*)(aBase0 + (kb + 16 + hf * 8) * 2);
            a1.q[0] = *(const int4*)(aBase1 + (kb + hf * 8) * 2);
            a1.q[1] = *(const int4*)(aBase1 + (kb + 16 + hf * 8) * 2);
#pragma unroll
            for (int j = 0; j < 4; ++j) {
                const int n = (ng * 4 + j) * 16 + rsub;
                const unsigned char* bp = w1tb + (size_t)n * (H * 2) + (kb + hf * 8) * 2;
                AFrag bf;                      // B layout: same K order as A (two 8-K runs)
                bf.q[0] = *(const int4*)(bp);
                bf.q[1] = *(const int4*)(bp + 32);
                acc[0][j] = __builtin_amdgcn_wmma_f32_16x16x32_bf16(
                              false, a0.v, false, bf.v, (short)0, acc[0][j], false, false);
                acc[1][j] = __builtin_amdgcn_wmma_f32_16x16x32_bf16(
                              false, a1.v, false, bf.v, (short)0, acc[1][j], false, false);
            }
        }
        // fold this N-block into per-row energy partials (branch-free tanh)
#pragma unroll
        for (int j = 0; j < 4; ++j) {
            const int col = (ng * 4 + j) * 16 + rsub;
            const float qv = s_qp[col];
            const float vv = s_v[col];
#pragma unroll
            for (int r = 0; r < 8; ++r) {
                e0[r] += vv * fast_tanh(acc[0][j][r] + qv);
                e1[r] += vv * fast_tanh(acc[1][j][r] + qv);
            }
        }
    }
    // reduce over the 16 columns held by lanes sharing this half
#pragma unroll
    for (int r = 0; r < 8; ++r) {
#pragma unroll
        for (int off = 8; off >= 1; off >>= 1) {
            e0[r] += __shfl_xor(e0[r], off, 32);
            e1[r] += __shfl_xor(e1[r], off, 32);
        }
    }
    if (rsub == 0) {                           // lanes 0 (rows 0-7) and 16 (rows 8-15)
        const int sb0 = srow0 + wave * 32 + hf * 8;
#pragma unroll
        for (int r = 0; r < 8; ++r) {
            energy[(size_t)b * SEQ + sb0 + r]      = e0[r];
            energy[(size_t)b * SEQ + sb0 + 16 + r] = e1[r];
        }
    }
}

// ---------------- K4: masked softmax over S per batch ----------------
__global__ void __launch_bounds__(256)
softmax_kernel(const float* __restrict__ energy, const int* __restrict__ lens,
               float* __restrict__ attn) {
    __shared__ float red[256];
    const int b = blockIdx.x, tid = threadIdx.x;
    const int len = lens[b];

    float mx = -3.0e38f;
    for (int s = tid; s < SEQ; s += 256) {
        float ev = (s < len) ? energy[b * SEQ + s] : NEG_INF;
        mx = fmaxf(mx, ev);
    }
    red[tid] = mx; __syncthreads();
    for (int o = 128; o > 0; o >>= 1) {
        if (tid < o) red[tid] = fmaxf(red[tid], red[tid + o]);
        __syncthreads();
    }
    mx = red[0]; __syncthreads();

    float sum = 0.f;
    for (int s = tid; s < SEQ; s += 256) {
        float ev = (s < len) ? energy[b * SEQ + s] : NEG_INF;
        sum += __expf(ev - mx);
    }
    red[tid] = sum; __syncthreads();
    for (int o = 128; o > 0; o >>= 1) {
        if (tid < o) red[tid] += red[tid + o];
        __syncthreads();
    }
    const float inv = 1.0f / red[0];

    for (int s = tid; s < SEQ; s += 256) {
        float ev = (s < len) ? energy[b * SEQ + s] : NEG_INF;
        attn[b * SEQ + s] = __expf(ev - mx) * inv;
    }
}

// ---------------- K5/K6: context = sum_s attn * enc (deterministic 2-stage) ----------------
__global__ void __launch_bounds__(512)
ctx_partial_kernel(const float* __restrict__ enc, const float* __restrict__ attn,
                   float* __restrict__ partial) {     // [B][8][H]
    const int b = blockIdx.x, c = blockIdx.y, h = threadIdx.x;
    const float* ep = enc + ((size_t)b * SEQ + c * 256) * H + h;
    const float* ap = attn + b * SEQ + c * 256;
    float acc = 0.f;
    for (int s = 0; s < 256; ++s) acc += ap[s] * ep[(size_t)s * H];
    partial[((size_t)b * 8 + c) * H + h] = acc;
}

__global__ void __launch_bounds__(512)
ctx_reduce_kernel(const float* __restrict__ partial, float* __restrict__ out) {
    const int b = blockIdx.x, h = threadIdx.x;
    float acc = 0.f;
#pragma unroll
    for (int c = 0; c < 8; ++c) acc += partial[((size_t)b * 8 + c) * H + h];
    out[(size_t)b * H + h] = acc;
}

// ---------------- launch ----------------
extern "C" void kernel_launch(void* const* d_in, const int* in_sizes, int n_in,
                              void* d_out, int out_size, void* d_ws, size_t ws_size,
                              hipStream_t stream) {
    const float* outp = (const float*)d_in[0];   // [B,1,H]
    const float* enc  = (const float*)d_in[1];   // [B,S,H]
    const int*   lens = (const int*)d_in[2];     // [B]
    const float* W1   = (const float*)d_in[3];   // [H,H]
    const float* W2   = (const float*)d_in[4];   // [H,H]
    const float* Vv   = (const float*)d_in[5];   // [H,1]
    float* out = (float*)d_out;                  // [B,H]

    char* ws = (char*)d_ws;
    unsigned short* w1t   = (unsigned short*)(ws + 0);            // 512 KB bf16
    float*          qproj = (float*)(ws + 524288);                // 128 KB
    float*          energy= (float*)(ws + 524288 + 131072);       // 512 KB
    float*          attn  = (float*)(ws + 524288 + 131072 + 524288);          // 512 KB
    float*          part  = (float*)(ws + 524288 + 131072 + 524288 + 524288); // 1 MB

    w1t_kernel   <<<(H * H) / 256, 256, 0, stream>>>(W1, w1t);
    qproj_kernel <<<BATCH, 256, 0, stream>>>(outp, W2, qproj);
    energy_kernel<<<BATCH * (SEQ / A_ROWS), 64, DYN_LDS_BYTES, stream>>>(enc, w1t, qproj, Vv, energy);
    softmax_kernel<<<BATCH, 256, 0, stream>>>(energy, lens, attn);
    ctx_partial_kernel<<<dim3(BATCH, 8), 512, 0, stream>>>(enc, attn, part);
    ctx_reduce_kernel <<<BATCH, 512, 0, stream>>>(part, out);
}
